// MotionGRU_7017976562228
// MI455X (gfx1250) — compile-verified, matrix operand order checked by reading, not run on hardware
//
#include <hip/hip_runtime.h>
#include <hip/hip_bf16.h>
#include <math.h>

// ---------------------------------------------------------------------------
// MotionGRU on gfx1250: all convs as bf16 WMMA GEMMs (v_wmma_f32_16x16x32_bf16)
// B-fragment tables staged per-workgroup into LDS (async DMA in k_warp).
// ---------------------------------------------------------------------------

typedef __attribute__((ext_vector_type(16))) __bf16 v16bf;
typedef __attribute__((ext_vector_type(8)))  float  v8f;

union Frag {
  v16bf v;
  uint4 u[2];
  unsigned short s[16];
};

#define BATCH 4
#define CCH   64
#define MHC   18
#define HH    128
#define WW    128
#define HP    130
#define WP    130

#define NG_FRAG (9 * 3 * 3 * 512)   // gates   B table (ushorts)
#define NC_FRAG (9 * 3 * 2 * 512)   // cand    B table
#define NM_FRAG (9 * 2 * 1 * 512)   // mod     B table
#define NW_FRAG (9 * 2 * 4 * 512)   // warp    B table

__device__ __forceinline__ float bf2f(unsigned short h) {
  union { unsigned u; float f; } c; c.u = ((unsigned)h) << 16; return c.f;
}
__device__ __forceinline__ unsigned short f2bf(float f) {
  union { float f; unsigned u; } c; c.f = f;
  unsigned r = c.u + 0x7FFFu + ((c.u >> 16) & 1u);
  return (unsigned short)(r >> 16);
}
__device__ __forceinline__ float sigmoidf(float x) { return 1.f / (1.f + expf(-x)); }

#define WMMA_BF16(A, B, C) \
  __builtin_amdgcn_wmma_f32_16x16x32_bf16(false, (A), false, (B), (short)0, (C), false, false)

// ------------------------- staging kernels ---------------------------------

// stacked [B][130][130][96] bf16: ch 0..63 = x_t, 64..81 = pre_offset, rest 0; zero halo
__global__ void k_stage_stk(const float* __restrict__ x, const float* __restrict__ pre,
                            unsigned short* __restrict__ stk) {
  int idx = blockIdx.x * blockDim.x + threadIdx.x;
  const int total = BATCH * HP * WP * 96;
  if (idx >= total) return;
  int c = idx % 96; int p = idx / 96;
  int wp = p % WP; p /= WP; int hp = p % HP; int b = p / HP;
  float v = 0.f;
  if (hp >= 1 && hp <= HH && wp >= 1 && wp <= WW) {
    int h = hp - 1, w = wp - 1;
    if (c < CCH)     v = x[((b * CCH + c) * HH + h) * WW + w];
    else if (c < 82) v = pre[((b * MHC + (c - 64)) * HH + h) * WW + w];
  }
  stk[idx] = f2bf(v);
}

// padded x [B][130][130][64] bf16 (channels-last for warp gathers)
__global__ void k_stage_xp(const float* __restrict__ x, unsigned short* __restrict__ xp) {
  int idx = blockIdx.x * blockDim.x + threadIdx.x;
  const int total = BATCH * HP * WP * CCH;
  if (idx >= total) return;
  int c = idx % CCH; int p = idx / CCH;
  int wp = p % WP; p /= WP; int hp = p % HP; int b = p / HP;
  float v = 0.f;
  if (hp >= 1 && hp <= HH && wp >= 1 && wp <= WW)
    v = x[((b * CCH + c) * HH + (hp - 1)) * WW + (wp - 1)];
  xp[idx] = f2bf(v);
}

// reset-modulated pre_offset halo buffer [B][130][130][32] bf16 -> zero fill
__global__ void k_zero_rmod(unsigned short* __restrict__ r) {
  int idx = blockIdx.x * blockDim.x + threadIdx.x;
  if (idx < BATCH * HP * WP * 32) r[idx] = 0;
}

// ------------------------- weight fragment prep ----------------------------
// B fragment storage: per 32x16 (KxN) tile, frag[lane][j] (16 bf16 / lane).
// lane<16: n=lane, k = j<8 ? j : j+8 ; lane>=16: n=lane-16, k = 8 + (j<8 ? j : j+8)
__global__ void k_prep_frags(const float* __restrict__ uw, const float* __restrict__ rw,
                             const float* __restrict__ ow, const float* __restrict__ gw,
                             const float* __restrict__ cw,
                             unsigned short* __restrict__ gB, unsigned short* __restrict__ cB,
                             unsigned short* __restrict__ mB, unsigned short* __restrict__ wB) {
  int idx = blockIdx.x * blockDim.x + threadIdx.x;
  if (idx >= NG_FRAG + NC_FRAG + NM_FRAG + NW_FRAG) return;
  int t = idx;
  unsigned short* dst;
  int region;
  if (t < NG_FRAG) { dst = gB; region = 0; }
  else if ((t -= NG_FRAG) < NC_FRAG) { dst = cB; region = 1; }
  else if ((t -= NC_FRAG) < NM_FRAG) { dst = mB; region = 2; }
  else { t -= NM_FRAG; dst = wB; region = 3; }
  int e = t % 512, f = t / 512;
  int lane = e / 16, j = e % 16, ln = lane & 15;
  int kc = ((lane >= 16) ? 8 : 0) + ((j < 8) ? j : j + 8);
  float v = 0.f;
  if (region == 0) {                       // gates: [tap][chunk3][nt3]
    int nt = f % 3, ch = (f / 3) % 3, tap = f / 9;
    int ci = ch * 32 + kc, co = nt * 16 + ln;
    if (ci < 82) {
      if (co < 18)      v = uw[(co * 82 + ci) * 9 + tap];
      else if (co < 36) v = rw[((co - 18) * 82 + ci) * 9 + tap];
    }
  } else if (region == 1) {                // cand: [tap][chunk3][nt2]
    int nt = f % 2, ch = (f / 2) % 3, tap = f / 6;
    int ci = ch * 32 + kc, co = nt * 16 + ln;
    if (ci < 82 && co < 18) v = ow[(co * 82 + ci) * 9 + tap];
  } else if (region == 2) {                // modulation: [tap][chunk2][nt1]
    int ch = f % 2, tap = f / 2;
    int ci = ch * 32 + kc, co = ln;
    if (co < 9) v = gw[(co * CCH + ci) * 9 + tap];
  } else {                                 // warp: [n][chunk2][nt4]
    int nt = f % 4, ch = (f / 4) % 2, n = f / 8;
    int ci = ch * 32 + kc, co = nt * 16 + ln;
    v = cw[(co * CCH + ci) * 9 + n];
  }
  dst[t] = f2bf(v);
}

// ------------------------- gate GEMM (upd + rst) ---------------------------
__global__ __launch_bounds__(256)
void k_gates(const unsigned short* __restrict__ stk, const unsigned short* __restrict__ gB,
             const float* __restrict__ ub, const float* __restrict__ rb,
             const float* __restrict__ pre,
             float* __restrict__ ug, unsigned short* __restrict__ rmod) {
  extern __shared__ unsigned short sB[];
  {
    const uint4* src = (const uint4*)gB;
    uint4* dst = (uint4*)sB;
    for (int i = threadIdx.x; i < NG_FRAG / 8; i += 256) dst[i] = src[i];
  }
  __syncthreads();
  const int wave = threadIdx.x >> 5, lane = threadIdx.x & 31;
  const int ln = lane & 15, hi = lane >> 4;
  const int b = blockIdx.x >> 7, h = blockIdx.x & 127;
  const int w0 = wave * 16, pw = w0 + ln;
  v8f a0 = {}, a1 = {}, a2 = {};
  #pragma unroll
  for (int tap = 0; tap < 9; ++tap) {
    const int dh = tap / 3, dw = tap % 3;
    const unsigned short* arow = stk + ((b * HP + h + dh) * WP + pw + dw) * 96 + hi * 8;
    if (tap < 8)
      __builtin_prefetch(stk + ((b * HP + h + (tap + 1) / 3) * WP + pw + (tap + 1) % 3) * 96, 0, 1);
    #pragma unroll
    for (int ch = 0; ch < 3; ++ch) {
      Frag a;
      a.u[0] = *(const uint4*)(arow + ch * 32);
      a.u[1] = *(const uint4*)(arow + ch * 32 + 16);
      const unsigned short* bb = sB + ((tap * 3 + ch) * 3) * 512 + lane * 16;
      Frag b0, b1, b2;
      b0.u[0] = *(const uint4*)(bb);        b0.u[1] = *(const uint4*)(bb + 8);
      b1.u[0] = *(const uint4*)(bb + 512);  b1.u[1] = *(const uint4*)(bb + 520);
      b2.u[0] = *(const uint4*)(bb + 1024); b2.u[1] = *(const uint4*)(bb + 1032);
      a0 = WMMA_BF16(a.v, b0.v, a0);
      a1 = WMMA_BF16(a.v, b1.v, a1);
      a2 = WMMA_BF16(a.v, b2.v, a2);
    }
  }
  #pragma unroll
  for (int r = 0; r < 8; ++r) {
    const int m = r + hi * 8, w = w0 + m;
    {                                         // nt0: channels 0..15 (update)
      const int n = ln;
      ug[((b * MHC + n) * HH + h) * WW + w] = sigmoidf(a0[r] + ub[n]);
    }
    {                                         // nt1: 16,17 update / 18..31 reset 0..13
      const int n = 16 + ln;
      if (n < 18) {
        ug[((b * MHC + n) * HH + h) * WW + w] = sigmoidf(a1[r] + ub[n]);
      } else {
        const int c = n - 18;
        const float s = sigmoidf(a1[r] + rb[c]);
        const float pv = pre[((b * MHC + c) * HH + h) * WW + w];
        rmod[((b * HP + h + 1) * WP + w + 1) * 32 + c] = f2bf(s * pv);
      }
    }
    {                                         // nt2: 32..35 -> reset 14..17
      const int n = 32 + ln;
      if (n < 36) {
        const int c = n - 18;
        const float s = sigmoidf(a2[r] + rb[c]);
        const float pv = pre[((b * MHC + c) * HH + h) * WW + w];
        rmod[((b * HP + h + 1) * WP + w + 1) * 32 + c] = f2bf(s * pv);
      }
    }
  }
}

// --------------------- candidate GEMM + GRU blend + mean -------------------
__global__ __launch_bounds__(256)
void k_cand(const unsigned short* __restrict__ stk, const unsigned short* __restrict__ rmod,
            const unsigned short* __restrict__ cB, const float* __restrict__ ob,
            const float* __restrict__ pre, const float* __restrict__ mean,
            const float* __restrict__ ug,
            float* __restrict__ off_out, float* __restrict__ mean_out) {
  extern __shared__ unsigned short sB[];
  {
    const uint4* src = (const uint4*)cB;
    uint4* dst = (uint4*)sB;
    for (int i = threadIdx.x; i < NC_FRAG / 8; i += 256) dst[i] = src[i];
  }
  __syncthreads();
  const int wave = threadIdx.x >> 5, lane = threadIdx.x & 31;
  const int ln = lane & 15, hi = lane >> 4;
  const int b = blockIdx.x >> 7, h = blockIdx.x & 127;
  const int w0 = wave * 16, pw = w0 + ln;
  v8f a0 = {}, a1 = {};
  #pragma unroll
  for (int tap = 0; tap < 9; ++tap) {
    const int dh = tap / 3, dw = tap % 3;
    const unsigned short* arow = stk + ((b * HP + h + dh) * WP + pw + dw) * 96 + hi * 8;
    const unsigned short* rrow = rmod + ((b * HP + h + dh) * WP + pw + dw) * 32 + hi * 8;
    #pragma unroll
    for (int ch = 0; ch < 3; ++ch) {
      Frag a;
      if (ch < 2) {
        a.u[0] = *(const uint4*)(arow + ch * 32);
        a.u[1] = *(const uint4*)(arow + ch * 32 + 16);
      } else {
        a.u[0] = *(const uint4*)(rrow);
        a.u[1] = *(const uint4*)(rrow + 16);
      }
      const unsigned short* bb = sB + ((tap * 3 + ch) * 2) * 512 + lane * 16;
      Frag b0, b1;
      b0.u[0] = *(const uint4*)(bb);       b0.u[1] = *(const uint4*)(bb + 8);
      b1.u[0] = *(const uint4*)(bb + 512); b1.u[1] = *(const uint4*)(bb + 520);
      a0 = WMMA_BF16(a.v, b0.v, a0);
      a1 = WMMA_BF16(a.v, b1.v, a1);
    }
  }
  #pragma unroll
  for (int r = 0; r < 8; ++r) {
    const int m = r + hi * 8, w = w0 + m;
    {
      const int n = ln;
      const int gi = ((b * MHC + n) * HH + h) * WW + w;
      const float cd = tanhf(a0[r] + ob[n]);
      const float u = ug[gi], po = pre[gi];
      const float mn = 0.5f * (mean[gi] + po);
      off_out[gi] = po * (1.f - u) + cd * u + mn;
      mean_out[gi] = mn;
    }
    {
      const int n = 16 + ln;
      if (n < 18) {
        const int gi = ((b * MHC + n) * HH + h) * WW + w;
        const float cd = tanhf(a1[r] + ob[n]);
        const float u = ug[gi], po = pre[gi];
        const float mn = 0.5f * (mean[gi] + po);
        off_out[gi] = po * (1.f - u) + cd * u + mn;
        mean_out[gi] = mn;
      }
    }
  }
}

// ------------------------- modulation GEMM (64 -> 9) -----------------------
__global__ __launch_bounds__(256)
void k_mod(const unsigned short* __restrict__ xp, const unsigned short* __restrict__ mB,
           const float* __restrict__ gb, float* __restrict__ mbuf) {
  extern __shared__ unsigned short sB[];
  {
    const uint4* src = (const uint4*)mB;
    uint4* dst = (uint4*)sB;
    for (int i = threadIdx.x; i < NM_FRAG / 8; i += 256) dst[i] = src[i];
  }
  __syncthreads();
  const int wave = threadIdx.x >> 5, lane = threadIdx.x & 31;
  const int ln = lane & 15, hi = lane >> 4;
  const int b = blockIdx.x >> 7, h = blockIdx.x & 127;
  const int w0 = wave * 16, pw = w0 + ln;
  v8f a0 = {};
  #pragma unroll
  for (int tap = 0; tap < 9; ++tap) {
    const int dh = tap / 3, dw = tap % 3;
    const unsigned short* arow = xp + ((b * HP + h + dh) * WP + pw + dw) * CCH + hi * 8;
    #pragma unroll
    for (int ch = 0; ch < 2; ++ch) {
      Frag a;
      a.u[0] = *(const uint4*)(arow + ch * 32);
      a.u[1] = *(const uint4*)(arow + ch * 32 + 16);
      const unsigned short* bb = sB + (tap * 2 + ch) * 512 + lane * 16;
      Frag b0;
      b0.u[0] = *(const uint4*)(bb); b0.u[1] = *(const uint4*)(bb + 8);
      a0 = WMMA_BF16(a.v, b0.v, a0);
    }
  }
  #pragma unroll
  for (int r = 0; r < 8; ++r) {
    const int n = ln;
    if (n < 9) {
      const int m = r + hi * 8, w = w0 + m;
      mbuf[((b * HH + h) * WW + w) * 9 + n] = sigmoidf(a0[r] + gb[n]);
    }
  }
}

// --------------- deformable warp: gather+bilinear fused into GEMM ----------
__global__ __launch_bounds__(256)
void k_warp(const unsigned short* __restrict__ xp, const unsigned short* __restrict__ wB,
            const float* __restrict__ off, const float* __restrict__ mbuf,
            float* __restrict__ xout) {
  extern __shared__ unsigned short sB[];
  // Async DMA of the 72KB warp weight-fragment table into LDS.
  // Dynamic-LDS base offset is 0 (no static __shared__ in this kernel), so the
  // LDS byte address equals the linear byte offset.
  {
    const unsigned tid = threadIdx.x;
    #pragma unroll
    for (int it = 0; it < (NW_FRAG * 2) / (256 * 16); ++it) {   // 18 iters
      unsigned byteoff = (it * 256 + tid) * 16;
      asm volatile("global_load_async_to_lds_b128 %0, %1, %2"
                   :: "v"(byteoff), "v"(byteoff), "s"(wB) : "memory");
    }
    asm volatile("s_wait_asynccnt 0x0" ::: "memory");
  }
  __syncthreads();
  const int wave = threadIdx.x >> 5, lane = threadIdx.x & 31;
  const int ln = lane & 15, hi = lane >> 4;
  const int b = blockIdx.x >> 7, h = blockIdx.x & 127;
  const int w0 = wave * 16, pw = w0 + ln;
  float offx[9], offy[9], mm[9];
  const float* op = off + (b * MHC * HH * WW) + h * WW + pw;
  const float* mp = mbuf + ((b * HH + h) * WW + pw) * 9;
  #pragma unroll
  for (int n = 0; n < 9; ++n) {
    offx[n] = op[n * HH * WW];
    offy[n] = op[(9 + n) * HH * WW];
    mm[n] = mp[n];
  }
  v8f a0 = {}, a1 = {}, a2 = {}, a3 = {};
  const unsigned short* xb = xp + b * (HP * WP * CCH);
  #pragma unroll
  for (int n = 0; n < 9; ++n) {
    const float rx = (float)(n / 3) - 1.f, ry = (float)(n % 3) - 1.f;
    float px = (float)(h + 1) + rx + offx[n];
    float py = (float)(pw + 1) + ry + offy[n];
    const float fx = floorf(px), fy = floorf(py);
    const float qxl = fminf(fmaxf(fx, 0.f), 129.f);
    const float qyl = fminf(fmaxf(fy, 0.f), 129.f);
    const float qxr = fminf(fmaxf(fx + 1.f, 0.f), 129.f);
    const float qyr = fminf(fmaxf(fy + 1.f, 0.f), 129.f);
    px = fminf(fmaxf(px, 0.f), 129.f);
    py = fminf(fmaxf(py, 0.f), 129.f);
    const float glt = (1.f + (qxl - px)) * (1.f + (qyl - py));
    const float grb = (1.f - (qxr - px)) * (1.f - (qyr - py));
    const float glb = (1.f + (qxl - px)) * (1.f - (qyr - py));
    const float grt = (1.f - (qxr - px)) * (1.f + (qyl - py));
    const int ixl = (int)qxl, iyl = (int)qyl, ixr = (int)qxr, iyr = (int)qyr;
    const unsigned short* plt = xb + (ixl * WP + iyl) * CCH + hi * 8;
    const unsigned short* prb = xb + (ixr * WP + iyr) * CCH + hi * 8;
    const unsigned short* plb = xb + (ixl * WP + iyr) * CCH + hi * 8;
    const unsigned short* prt = xb + (ixr * WP + iyl) * CCH + hi * 8;
    const float mn = mm[n];
    #pragma unroll
    for (int ch = 0; ch < 2; ++ch) {
      Frag a;
      #pragma unroll
      for (int hf = 0; hf < 2; ++hf) {
        const int co = ch * 32 + hf * 16;
        union { uint4 q; unsigned short s[8]; } clt, crb, clb, crt;
        clt.q = *(const uint4*)(plt + co);
        crb.q = *(const uint4*)(prb + co);
        clb.q = *(const uint4*)(plb + co);
        crt.q = *(const uint4*)(prt + co);
        #pragma unroll
        for (int j = 0; j < 8; ++j) {
          const float v = glt * bf2f(clt.s[j]) + grb * bf2f(crb.s[j])
                        + glb * bf2f(clb.s[j]) + grt * bf2f(crt.s[j]);
          a.s[hf * 8 + j] = f2bf(v * mn);
        }
      }
      const unsigned short* bb = sB + ((n * 2 + ch) * 4) * 512 + lane * 16;
      Frag b0, b1, b2, b3;
      b0.u[0] = *(const uint4*)(bb);        b0.u[1] = *(const uint4*)(bb + 8);
      b1.u[0] = *(const uint4*)(bb + 512);  b1.u[1] = *(const uint4*)(bb + 520);
      b2.u[0] = *(const uint4*)(bb + 1024); b2.u[1] = *(const uint4*)(bb + 1032);
      b3.u[0] = *(const uint4*)(bb + 1536); b3.u[1] = *(const uint4*)(bb + 1544);
      a0 = WMMA_BF16(a.v, b0.v, a0);
      a1 = WMMA_BF16(a.v, b1.v, a1);
      a2 = WMMA_BF16(a.v, b2.v, a2);
      a3 = WMMA_BF16(a.v, b3.v, a3);
    }
  }
  #pragma unroll
  for (int r = 0; r < 8; ++r) {
    const int m = r + hi * 8, w = w0 + m;
    xout[((b * CCH + (ln)) * HH + h) * WW + w] = a0[r];
    xout[((b * CCH + (16 + ln)) * HH + h) * WW + w] = a1[r];
    xout[((b * CCH + (32 + ln)) * HH + h) * WW + w] = a2[r];
    xout[((b * CCH + (48 + ln)) * HH + h) * WW + w] = a3[r];
  }
}

// ------------------------------ launch -------------------------------------
extern "C" void kernel_launch(void* const* d_in, const int* in_sizes, int n_in,
                              void* d_out, int out_size, void* d_ws, size_t ws_size,
                              hipStream_t stream) {
  (void)in_sizes; (void)n_in; (void)out_size; (void)ws_size;
  const float* x_t   = (const float*)d_in[0];
  const float* pre   = (const float*)d_in[1];
  const float* mean  = (const float*)d_in[2];
  const float* upd_w = (const float*)d_in[3];
  const float* upd_b = (const float*)d_in[4];
  const float* rst_w = (const float*)d_in[5];
  const float* rst_b = (const float*)d_in[6];
  const float* out_w = (const float*)d_in[7];
  const float* out_b = (const float*)d_in[8];
  const float* wg_w  = (const float*)d_in[9];
  const float* wg_b  = (const float*)d_in[10];
  const float* wcw   = (const float*)d_in[11];

  char* ws = (char*)d_ws;
  unsigned short* stk  = (unsigned short*)(ws);             // 4*130*130*96*2  = 12,979,200
  unsigned short* xpb  = (unsigned short*)(ws + 12979200);  // 4*130*130*64*2  =  8,652,800
  unsigned short* rmod = (unsigned short*)(ws + 21632000);  // 4*130*130*32*2  =  4,326,400
  float* ug            = (float*)(ws + 25958400);           // 4*18*128*128*4  =  4,718,592
  float* mbuf          = (float*)(ws + 30676992);           // 4*128*128*9*4   =  2,359,296
  unsigned short* gB   = (unsigned short*)(ws + 33036288);  // 81 frags        =     82,944
  unsigned short* cB   = (unsigned short*)(ws + 33119232);  // 54 frags        =     55,296
  unsigned short* mB   = (unsigned short*)(ws + 33174528);  // 18 frags        =     18,432
  unsigned short* wBf  = (unsigned short*)(ws + 33192960);  // 72 frags        =     73,728

  float* outp     = (float*)d_out;
  float* x_out    = outp;            // [4,64,128,128]
  float* off_out  = outp + 4194304;  // [4,18,128,128]
  float* mean_out = outp + 5373952;  // [4,18,128,128]

  dim3 blk(256);
  k_stage_stk<<<(BATCH * HP * WP * 96 + 255) / 256, blk, 0, stream>>>(x_t, pre, stk);
  k_stage_xp<<<(BATCH * HP * WP * CCH + 255) / 256, blk, 0, stream>>>(x_t, xpb);
  k_zero_rmod<<<(BATCH * HP * WP * 32 + 255) / 256, blk, 0, stream>>>(rmod);
  k_prep_frags<<<(NG_FRAG + NC_FRAG + NM_FRAG + NW_FRAG + 255) / 256, blk, 0, stream>>>(
      upd_w, rst_w, out_w, wg_w, wcw, gB, cB, mB, wBf);
  k_gates<<<BATCH * HH, blk, NG_FRAG * 2, stream>>>(stk, gB, upd_b, rst_b, pre, ug, rmod);
  k_cand<<<BATCH * HH, blk, NC_FRAG * 2, stream>>>(stk, rmod, cB, out_b, pre, mean, ug,
                                                   off_out, mean_out);
  k_mod<<<BATCH * HH, blk, NM_FRAG * 2, stream>>>(xpb, mB, wg_b, mbuf);
  k_warp<<<BATCH * HH, blk, NW_FRAG * 2, stream>>>(xpb, wBf, off_out, mbuf, x_out);
}
